// LSTM_1142461300796
// MI455X (gfx1250) — matrix operand order repaired and checked by reading, not compiled, not run on hardware
//
#include <hip/hip_runtime.h>
#include <hip/hip_bf16.h>

#define B_   64
#define S_   512
#define H_   1024
#define IN_  1024
#define G4_  4096   // 4*H

typedef __attribute__((ext_vector_type(16))) __bf16 v16bf;
typedef __attribute__((ext_vector_type(8)))  float  v8f;

__device__ __forceinline__ float sigmoidf_(float x) {
    return 1.0f / (1.0f + __expf(-x));
}

// ---- WMMA fragment loaders (CDNA5 16x16x32 bf16 layouts, ISA 7.12.2) ----
// A: 16x32 (MxK). lanes 0-15: M=lane, K 0..7 / 16..23; lanes 16-31: K 8..15 / 24..31.
__device__ __forceinline__ v16bf load_a16x32(const __bf16* base, int ld, int lane) {
    const int row = lane & 15;
    const int kb  = (lane & 16) ? 8 : 0;
    const __bf16* p = base + row * ld + kb;
    v16bf a;
#pragma unroll
    for (int j = 0; j < 8; ++j) { a[j] = p[j]; a[8 + j] = p[16 + j]; }
    return a;
}

// B: 32x16 (KxN), one column per lane; lanes 0-15 K 0..7 / 16..23; lanes 16-31: K 8..15 / 24..31.
// Memory holds W as (N, K) row-major, so a lane reads two runs of 8 contiguous bf16 (at +0 and +16).
__device__ __forceinline__ v16bf load_b32x16(const __bf16* wbase, int ld, int lane) {
    const int col = lane & 15;
    const int ko  = (lane & 16) ? 8 : 0;
    const __bf16* p = wbase + col * ld + ko;
    v16bf b;
#pragma unroll
    for (int j = 0; j < 8; ++j) { b[j] = p[j]; b[8 + j] = p[16 + j]; }
    return b;
}

#define WMMA_BF16(a, b, c) \
    __builtin_amdgcn_wmma_f32_16x16x32_bf16(false, (a), false, (b), (short)0, (c), false, false)

// Async global -> LDS copy of 16B per lane (ASYNCcnt-tracked, bypasses VGPRs).
// The LDS destination is passed as the REAL generic pointer: its low 32 bits are
// the wave-relative LDS byte address (what VDST expects), and the ptrtoint makes
// the shared allocation escape so subsequent ds_loads are not folded away.
__device__ __forceinline__ void async_g2l_b128(void* lds, const void* gaddr) {
    asm volatile("global_load_async_to_lds_b128 %0, %1, off"
                 :: "v"((unsigned)(unsigned long long)lds), "v"(gaddr) : "memory");
}
__device__ __forceinline__ void async_g2l_b128_o16(void* lds, const void* gaddr) {
    asm volatile("global_load_async_to_lds_b128 %0, %1, off offset:16"
                 :: "v"((unsigned)(unsigned long long)lds), "v"(gaddr) : "memory");
}
__device__ __forceinline__ void async_g2l_b128_o32(void* lds, const void* gaddr) {
    asm volatile("global_load_async_to_lds_b128 %0, %1, off offset:32"
                 :: "v"((unsigned)(unsigned long long)lds), "v"(gaddr) : "memory");
}
__device__ __forceinline__ void async_g2l_b128_o48(void* lds, const void* gaddr) {
    asm volatile("global_load_async_to_lds_b128 %0, %1, off offset:48"
                 :: "v"((unsigned)(unsigned long long)lds), "v"(gaddr) : "memory");
}

// ---------------- packing / init kernels ----------------
__global__ void pack_weights(const float* __restrict__ wii, const float* __restrict__ wif,
                             const float* __restrict__ wig, const float* __restrict__ wio,
                             const float* __restrict__ whi, const float* __restrict__ whf,
                             const float* __restrict__ whg, const float* __restrict__ who,
                             __bf16* __restrict__ Wi, __bf16* __restrict__ Wh) {
    size_t idx = (size_t)blockIdx.x * blockDim.x + threadIdx.x;
    if (idx >= (size_t)G4_ * IN_) return;
    const int    gate = (int)(idx / ((size_t)H_ * IN_));
    const size_t e    = idx % ((size_t)H_ * IN_);
    const float* wi = gate == 0 ? wii : gate == 1 ? wif : gate == 2 ? wig : wio;
    const float* wh = gate == 0 ? whi : gate == 1 ? whf : gate == 2 ? whg : who;
    Wi[idx] = (__bf16)wi[e];
    Wh[idx] = (__bf16)wh[e];
}

__global__ void pack_bias(const float* __restrict__ bii, const float* __restrict__ bif,
                          const float* __restrict__ big, const float* __restrict__ bio,
                          const float* __restrict__ bhi, const float* __restrict__ bhf,
                          const float* __restrict__ bhg, const float* __restrict__ bho,
                          float* __restrict__ bias) {
    int n = blockIdx.x * blockDim.x + threadIdx.x;
    if (n >= G4_) return;
    const int gate = n / H_, j = n % H_;
    const float* bi = gate == 0 ? bii : gate == 1 ? bif : gate == 2 ? big : bio;
    const float* bh = gate == 0 ? bhi : gate == 1 ? bhf : gate == 2 ? bhg : bho;
    bias[n] = bi[j] + bh[j];
}

// inputs (B,S,IN) f32 -> Xt (S*B, IN) bf16 with m = s*B + b
__global__ void pack_inputs(const float* __restrict__ x, __bf16* __restrict__ Xt) {
    size_t idx = (size_t)blockIdx.x * blockDim.x + threadIdx.x;
    if (idx >= (size_t)B_ * S_ * IN_) return;
    const size_t m = idx / IN_, k = idx % IN_;
    const int b = (int)(m % B_), s = (int)(m / B_);
    Xt[idx] = (__bf16)x[((size_t)b * S_ + s) * IN_ + k];
}

__global__ void init_state(const float* __restrict__ h0, const float* __restrict__ c0,
                           __bf16* __restrict__ hbf, float* __restrict__ c) {
    int i = blockIdx.x * blockDim.x + threadIdx.x;
    if (i >= B_ * H_) return;
    hbf[i] = (__bf16)h0[i];
    c[i]   = c0[i];
}

// ---------------- x_proj GEMM: (32768 x 1024) * (1024 x 4096)^T ----------------
// Block tile 64(M) x 256(N), 8 waves of 32x64; K panels of 32, async-staged into
// ping-pong LDS buffers (software pipeline on ASYNCcnt).
#define PANEL_A 2048                  // 64*32 bf16
#define PANEL_B 8192                  // 256*32 bf16
#define PANEL   (PANEL_A + PANEL_B)   // 10240 bf16 = 20 KB

__global__ __launch_bounds__(256)
void xproj_gemm(const __bf16* __restrict__ X,    // (M, K) row-major
                const __bf16* __restrict__ W,    // (N, K) row-major
                const float*  __restrict__ bias, // (N)
                float* __restrict__ out) {       // (M, N)
    __shared__ __align__(16) __bf16 smem[2 * PANEL];  // 40 KB ping-pong

    const int tid  = threadIdx.x;
    const int lane = tid & 31;
    const int wave = tid >> 5;          // 0..7
    const int wm   = (wave & 1) * 32;   // 0 / 32
    const int wn   = (wave >> 1) * 64;  // 0 / 64 / 128 / 192
    const int blockM = blockIdx.x * 64;
    const int blockN = blockIdx.y * 256;

    // Per-thread staging assignment: A = one 16B chunk, B = one 64B row.
    const int arow = tid >> 2, ach = (tid & 3) * 8;  // A: 64 rows x 4 chunks
    const int brow = tid;                            // B: 256 rows x 64 B

    v8f acc[2][4] = {};

    auto stage = [&](int buf, int k0) {
        __bf16* adst = smem + buf * PANEL + arow * 32 + ach;
        async_g2l_b128(adst, &X[(size_t)(blockM + arow) * IN_ + k0 + ach]);
        __bf16* bdst = smem + buf * PANEL + PANEL_A + brow * 32;
        const __bf16* bg = &W[(size_t)(blockN + brow) * IN_ + k0];
        async_g2l_b128(bdst, bg);
        async_g2l_b128_o16(bdst, bg);
        async_g2l_b128_o32(bdst, bg);
        async_g2l_b128_o48(bdst, bg);
    };

    stage(0, 0);
    int buf = 0;
    for (int k0 = 0; k0 < IN_; k0 += 32) {
        if (k0 + 32 < IN_) {
            stage(buf ^ 1, k0 + 32);                           // prefetch next panel
            asm volatile("s_wait_asynccnt 0x5" ::: "memory");  // current panel landed
        } else {
            asm volatile("s_wait_asynccnt 0x0" ::: "memory");
        }
        __syncthreads();   // panel visible to all waves

        const __bf16* As = smem + buf * PANEL;
        const __bf16* Bs = As + PANEL_A;
        const v16bf a0 = load_a16x32(As + (wm + 0)  * 32, 32, lane);
        const v16bf a1 = load_a16x32(As + (wm + 16) * 32, 32, lane);
#pragma unroll
        for (int tn = 0; tn < 4; ++tn) {
            const v16bf b = load_b32x16(Bs + (wn + tn * 16) * 32, 32, lane);
            acc[0][tn] = WMMA_BF16(a0, b, acc[0][tn]);
            acc[1][tn] = WMMA_BF16(a1, b, acc[1][tn]);
        }
        __syncthreads();   // all reads done before this buffer is re-staged
        buf ^= 1;
    }

    const int lrow = (lane & 16) ? 8 : 0;
    const int lcol = lane & 15;
#pragma unroll
    for (int tm = 0; tm < 2; ++tm) {
#pragma unroll
        for (int tn = 0; tn < 4; ++tn) {
            const int gm = blockM + wm + tm * 16 + lrow;
            const int gn = blockN + wn + tn * 16 + lcol;
            const float bv = bias[gn];
#pragma unroll
            for (int r = 0; r < 8; ++r)
                out[(size_t)(gm + r) * G4_ + gn] = acc[tm][tn][r] + bv;
        }
    }
}

// ---------------- recurrent step ----------------
// Each wave owns one 16x16 (m, j) tile and all 4 gates at that tile.
__global__ __launch_bounds__(128)
void lstm_step(const float*  __restrict__ xp_t,    // (B, 4H), bias folded in
               const __bf16* __restrict__ Wh,      // (4H, H) row-major
               const __bf16* __restrict__ h_in,    // (B, H) bf16
               __bf16*       __restrict__ h_out,   // (B, H) bf16
               float*        __restrict__ c_state, // (B, H) f32
               float*        __restrict__ outputs, // (B, S, H) f32
               int t) {
    const int lane = threadIdx.x & 31;
    const int wave = threadIdx.x >> 5;
    const int tile = blockIdx.x * 4 + wave;  // 0..255
    const int mt   = (tile >> 6) * 16;       // batch-tile base
    const int jt   = (tile & 63) * 16;       // hidden-tile base

    const int lrow = (lane & 16) ? 8 : 0;
    const int lcol = lane & 15;

    v8f ai = {}, af = {}, ag = {}, ao = {};
#pragma unroll
    for (int r = 0; r < 8; ++r) {
        const size_t m = (size_t)(mt + lrow + r);
        ai[r] = xp_t[m * G4_ + 0 * H_ + jt + lcol];
        af[r] = xp_t[m * G4_ + 1 * H_ + jt + lcol];
        ag[r] = xp_t[m * G4_ + 2 * H_ + jt + lcol];
        ao[r] = xp_t[m * G4_ + 3 * H_ + jt + lcol];
    }

    const __bf16* wb0 = Wh + (size_t)(0 * H_ + jt) * H_;
    const __bf16* wb1 = Wh + (size_t)(1 * H_ + jt) * H_;
    const __bf16* wb2 = Wh + (size_t)(2 * H_ + jt) * H_;
    const __bf16* wb3 = Wh + (size_t)(3 * H_ + jt) * H_;

    for (int k0 = 0; k0 < H_; k0 += 32) {
        if (k0 + 32 < H_) {   // L2-resident weights: pull next panel toward L0
            __builtin_prefetch(wb0 + (size_t)(lane & 15) * H_ + k0 + 32, 0, 1);
            __builtin_prefetch(wb1 + (size_t)(lane & 15) * H_ + k0 + 32, 0, 1);
            __builtin_prefetch(wb2 + (size_t)(lane & 15) * H_ + k0 + 32, 0, 1);
            __builtin_prefetch(wb3 + (size_t)(lane & 15) * H_ + k0 + 32, 0, 1);
        }
        const v16bf a   = load_a16x32(h_in + (size_t)mt * H_ + k0, H_, lane);
        const v16bf bi  = load_b32x16(wb0 + k0, H_, lane);
        const v16bf bff = load_b32x16(wb1 + k0, H_, lane);
        const v16bf bg  = load_b32x16(wb2 + k0, H_, lane);
        const v16bf bo  = load_b32x16(wb3 + k0, H_, lane);
        ai = WMMA_BF16(a, bi,  ai);
        af = WMMA_BF16(a, bff, af);
        ag = WMMA_BF16(a, bg,  ag);
        ao = WMMA_BF16(a, bo,  ao);
    }

#pragma unroll
    for (int r = 0; r < 8; ++r) {
        const int m = mt + lrow + r;
        const int j = jt + lcol;
        const float gi = sigmoidf_(ai[r]);
        const float gf = sigmoidf_(af[r]);
        const float gg = tanhf(ag[r]);
        const float go = sigmoidf_(ao[r]);
        const size_t cidx = (size_t)m * H_ + j;
        const float cn = gf * c_state[cidx] + gi * gg;
        c_state[cidx] = cn;
        const float hn = go * tanhf(cn);
        h_out[cidx] = (__bf16)hn;
        outputs[((size_t)m * S_ + t) * H_ + j] = hn;
    }
}

__global__ void finalize(const float* __restrict__ outputs, const float* __restrict__ c_state,
                         float* __restrict__ hf, float* __restrict__ cf) {
    int i = blockIdx.x * blockDim.x + threadIdx.x;
    if (i >= B_ * H_) return;
    const int m = i / H_, j = i % H_;
    hf[i] = outputs[((size_t)m * S_ + (S_ - 1)) * H_ + j];
    cf[i] = c_state[i];
}

// ---------------- host launcher ----------------
extern "C" void kernel_launch(void* const* d_in, const int* in_sizes, int n_in,
                              void* d_out, int out_size, void* d_ws, size_t ws_size,
                              hipStream_t stream) {
    const float* inputs = (const float*)d_in[0];
    const float* h0     = (const float*)d_in[1];
    const float* c0     = (const float*)d_in[2];
    const float* w_ii = (const float*)d_in[3];
    const float* w_if = (const float*)d_in[4];
    const float* w_ig = (const float*)d_in[5];
    const float* w_io = (const float*)d_in[6];
    const float* b_ii = (const float*)d_in[7];
    const float* b_if = (const float*)d_in[8];
    const float* b_ig = (const float*)d_in[9];
    const float* b_io = (const float*)d_in[10];
    const float* w_hi = (const float*)d_in[11];
    const float* w_hf = (const float*)d_in[12];
    const float* w_hg = (const float*)d_in[13];
    const float* w_ho = (const float*)d_in[14];
    const float* b_hi = (const float*)d_in[15];
    const float* b_hf = (const float*)d_in[16];
    const float* b_hg = (const float*)d_in[17];
    const float* b_ho = (const float*)d_in[18];
    float* out = (float*)d_out;

    char* p = (char*)d_ws;
    __bf16* Wi   = (__bf16*)p; p += (size_t)G4_ * IN_ * sizeof(__bf16);
    __bf16* Wh   = (__bf16*)p; p += (size_t)G4_ * IN_ * sizeof(__bf16);
    float*  bias = (float*)p;  p += (size_t)G4_ * sizeof(float);
    __bf16* Xt   = (__bf16*)p; p += (size_t)B_ * S_ * IN_ * sizeof(__bf16);
    float*  xp   = (float*)p;  p += (size_t)B_ * S_ * G4_ * sizeof(float);
    __bf16* hbuf0 = (__bf16*)p; p += (size_t)B_ * H_ * sizeof(__bf16);
    __bf16* hbuf1 = (__bf16*)p; p += (size_t)B_ * H_ * sizeof(__bf16);
    float*  c_st  = (float*)p;  p += (size_t)B_ * H_ * sizeof(float);

    pack_weights<<<(G4_ * IN_ + 255) / 256, 256, 0, stream>>>(
        w_ii, w_if, w_ig, w_io, w_hi, w_hf, w_hg, w_ho, Wi, Wh);
    pack_bias<<<(G4_ + 255) / 256, 256, 0, stream>>>(
        b_ii, b_if, b_ig, b_io, b_hi, b_hf, b_hg, b_ho, bias);
    {
        size_t n = (size_t)B_ * S_ * IN_;
        pack_inputs<<<(unsigned)((n + 255) / 256), 256, 0, stream>>>(inputs, Xt);
    }
    init_state<<<(B_ * H_ + 255) / 256, 256, 0, stream>>>(h0, c0, hbuf0, c_st);

    dim3 grid((B_ * S_) / 64, G4_ / 256);
    xproj_gemm<<<grid, 256, 0, stream>>>(Xt, Wi, bias, xp);

    for (int t = 0; t < S_; ++t) {
        const __bf16* hin = (t & 1) ? hbuf1 : hbuf0;
        __bf16*      hout = (t & 1) ? hbuf0 : hbuf1;
        lstm_step<<<64, 128, 0, stream>>>(xp + (size_t)t * B_ * G4_, Wh, hin, hout,
                                          c_st, out, t);
    }

    finalize<<<(B_ * H_ + 255) / 256, 256, 0, stream>>>(
        out, c_st, out + (size_t)B_ * S_ * H_, out + (size_t)B_ * S_ * H_ + (size_t)B_ * H_);
}
